// CausalLinearAttention_4045859193553
// MI455X (gfx1250) — matrix operand order repaired and checked by reading, not compiled, not run on hardware
//
#include <hip/hip_runtime.h>

typedef __attribute__((ext_vector_type(16))) _Float16 v16h;
typedef __attribute__((ext_vector_type(8)))  _Float16 v8h;
typedef __attribute__((ext_vector_type(8)))  float    v8f;

#define NB     4
#define LSEQ   4096
#define NH     8
#define EDIM   64
#define CC     64            // chunk size (impl detail; math identical to reference)
#define NCHUNK (LSEQ / CC)
#define EPS_D  1e-6f

__device__ __forceinline__ float featmap(float x) {
    // elu(x)+1 == x+1 (x>0) else exp(x)
    return x > 0.f ? x + 1.f : __expf(x);
}

// ---- WMMA fragment helpers (CDNA5 16x16x32 f16, wave32 layouts, ISA 7.12.2) ----

// A tile: 16 rows x 32 K, from row-major src (k contiguous). One contiguous 32B run per lane.
__device__ __forceinline__ v16h load_A(const _Float16* src, int ld, int row0, int k0) {
    const int lane = threadIdx.x & 31;
    const int r    = lane & 15;
    const int kb   = (lane >> 4) << 4;     // 0 or 16
    const _Float16* p = src + (row0 + r) * ld + k0 + kb;
    return *(const v16h*)p;                // K kb..kb+15
}

// B tile: 32 K x 16 cols from a K-CONTIGUOUS (transposed) store: src[n][k], ld=ldk.
// Lane's 16 halves are one contiguous 32B run -> 2x ds_load_b128.
__device__ __forceinline__ v16h load_Bt(const _Float16* src, int ldk, int k0, int n0) {
    const int lane = threadIdx.x & 31;
    const int n    = lane & 15;
    const int kb   = (lane >> 4) << 4;     // 0 or 16
    const _Float16* p = src + (n0 + n) * ldk + k0 + kb;
    return *(const v16h*)p;
}

// Store C/D accum as f16 row-major (strided per v; used for attn only).
__device__ __forceinline__ void store_C_f16(_Float16* dst, int ld, int row0, int col0, v8f c) {
    const int lane = threadIdx.x & 31;
    const int n    = lane & 15;
    const int mb   = (lane >> 4) << 3;
#pragma unroll
    for (int v = 0; v < 8; ++v)
        dst[(row0 + mb + v) * ld + col0 + n] = (_Float16)c[v];
}

// Store C/D accum TRANSPOSED as f16: dst[col][row], contiguous in v -> 1x ds_store_b128.
__device__ __forceinline__ void store_C_trans_f16(_Float16* dst, int ld, int row0, int col0, v8f c) {
    const int lane = threadIdx.x & 31;
    const int n    = lane & 15;
    const int mb   = (lane >> 4) << 3;
    v8h h;
#pragma unroll
    for (int v = 0; v < 8; ++v) h[v] = (_Float16)c[v];
    *(v8h*)(dst + (col0 + n) * ld + row0 + mb) = h;
}

__global__ __launch_bounds__(128)
void causal_linear_attn_kernel(const float* __restrict__ Qg,
                               const float* __restrict__ Kg,
                               const float* __restrict__ Vg,
                               const float* __restrict__ Lm,
                               float* __restrict__ Og) {
    __shared__ _Float16 Qs [CC * EDIM];    // phi(Q), [l][e]        (A: attn, inter)
    __shared__ _Float16 Ks [CC * EDIM];    // phi(K), [s][e]        (B: attn, k=e contiguous)
    __shared__ _Float16 Kts[EDIM * CC];    // phi(K)^T, [e][s]      (A: S-update; ksum rows)
    __shared__ _Float16 Vts[EDIM * CC];    // V^T, [m][s]           (B: intra + S-update, k=s contiguous)
    __shared__ _Float16 As [CC * CC];      // masked attn, [l][s]   (A: intra; denom rows)
    __shared__ _Float16 Sft[EDIM * EDIM];  // S^T snapshot, [m][e]  (B: inter, k=e contiguous)
    __shared__ float    ksum[EDIM];
    __shared__ float    zbuf[CC];

    const int tid   = threadIdx.x;
    const int wv    = tid >> 5;
    const int lane  = tid & 31;
    const int nlane = lane & 15;
    const int mb    = (lane >> 4) << 3;
    const int nb    = blockIdx.x / NH;
    const int hh    = blockIdx.x % NH;

    v8f Sreg[4] = {};                      // S band rows e=[16wv,16wv+16), 4 m-tiles
    if (tid < EDIM) ksum[tid] = 0.f;
    __syncthreads();

    for (int c = 0; c < NCHUNK; ++c) {
        const int l0 = c * CC;

        // (1) publish pre-chunk S as S^T (contiguous b128 stores)
#pragma unroll
        for (int j = 0; j < 4; ++j)
            store_C_trans_f16(Sft, EDIM, wv * 16, j * 16, Sreg[j]);
        __syncthreads();

        // (2) stage + feature-map chunk (float4 global loads; prefetch next chunk)
        {
            const int r  = tid >> 1;
            const int e0 = (tid & 1) * 32;
            const size_t base = ((size_t)(nb * LSEQ + l0 + r) * NH + hh) * EDIM + e0;
            const float kl = Lm[(size_t)nb * LSEQ + l0 + r];
            const float4* q4 = (const float4*)(Qg + base);
            const float4* k4 = (const float4*)(Kg + base);
            const float4* v4 = (const float4*)(Vg + base);
            if (c + 1 < NCHUNK) {
                const size_t nxt = base + (size_t)CC * NH * EDIM;
                __builtin_prefetch(Qg + nxt, 0, 1);
                __builtin_prefetch(Kg + nxt, 0, 1);
                __builtin_prefetch(Vg + nxt, 0, 1);
            }
#pragma unroll
            for (int i = 0; i < 4; ++i) {
                float4 a0 = q4[2*i], a1 = q4[2*i+1];
                v8h qh = { (_Float16)featmap(a0.x), (_Float16)featmap(a0.y),
                           (_Float16)featmap(a0.z), (_Float16)featmap(a0.w),
                           (_Float16)featmap(a1.x), (_Float16)featmap(a1.y),
                           (_Float16)featmap(a1.z), (_Float16)featmap(a1.w) };
                *(v8h*)(Qs + r * EDIM + e0 + i * 8) = qh;

                float4 b0 = k4[2*i], b1 = k4[2*i+1];
                v8h kh = { (_Float16)(featmap(b0.x)*kl), (_Float16)(featmap(b0.y)*kl),
                           (_Float16)(featmap(b0.z)*kl), (_Float16)(featmap(b0.w)*kl),
                           (_Float16)(featmap(b1.x)*kl), (_Float16)(featmap(b1.y)*kl),
                           (_Float16)(featmap(b1.z)*kl), (_Float16)(featmap(b1.w)*kl) };
                *(v8h*)(Ks + r * EDIM + e0 + i * 8) = kh;
#pragma unroll
                for (int t = 0; t < 8; ++t) Kts[(e0 + i*8 + t) * CC + r] = kh[t];

                float4 c0 = v4[2*i], c1 = v4[2*i+1];
                v8h vh = { (_Float16)c0.x, (_Float16)c0.y, (_Float16)c0.z, (_Float16)c0.w,
                           (_Float16)c1.x, (_Float16)c1.y, (_Float16)c1.z, (_Float16)c1.w };
#pragma unroll
                for (int t = 0; t < 8; ++t) Vts[(e0 + i*8 + t) * CC + r] = vh[t];
            }
        }
        __syncthreads();

        // Q band A-fragments: shared by attn (3) and inter (5)
        const v16h qA0 = load_A(Qs, EDIM, wv * 16, 0);
        const v16h qA1 = load_A(Qs, EDIM, wv * 16, 32);

        // (3) attn band = Q[band] x K^T, causal-masked -> As
#pragma unroll
        for (int j = 0; j < 4; ++j) {
            v8f acc = {};
            if (j <= wv) {                 // tiles above diagonal are all-zero: skip
                acc = __builtin_amdgcn_wmma_f32_16x16x32_f16(
                          false, qA0, false, load_Bt(Ks, EDIM, 0,  j * 16), (short)0, acc, false, false);
                acc = __builtin_amdgcn_wmma_f32_16x16x32_f16(
                          false, qA1, false, load_Bt(Ks, EDIM, 32, j * 16), (short)0, acc, false, false);
                if (j == wv) {             // diagonal tile: keep col<=row
#pragma unroll
                    for (int v = 0; v < 8; ++v)
                        if (j * 16 + nlane > wv * 16 + mb + v) acc[v] = 0.f;
                }
            }
            store_C_f16(As, CC, wv * 16, j * 16, acc);
        }
        __syncthreads();

        // (4a) denom per row: rowsum(masked attn) + Q[l].ksum_prev  (b128 LDS reads)
        if (tid < CC) {
            float d = 0.f;
            const v8h* ar = (const v8h*)(As + tid * CC);
            const v8h* qr = (const v8h*)(Qs + tid * EDIM);
#pragma unroll
            for (int i = 0; i < 8; ++i) {
                v8h a = ar[i], q = qr[i];
#pragma unroll
                for (int t = 0; t < 8; ++t) {
                    d += (float)a[t];
                    d += (float)q[t] * ksum[i * 8 + t];
                }
            }
            zbuf[tid] = 1.f / (d + EPS_D);
        }
        __syncthreads();
        // (4b) fold chunk's K column-sums into ksum (consumed next chunk)
        if (tid < EDIM) {
            float s2 = ksum[tid];
            const v8h* kr = (const v8h*)(Kts + tid * CC);
#pragma unroll
            for (int i = 0; i < 8; ++i) {
                v8h kx = kr[i];
#pragma unroll
                for (int t = 0; t < 8; ++t) s2 += (float)kx[t];
            }
            ksum[tid] = s2;
        }

        // (5)+(6) merged: out = Q x S + attn x V (scaled), and S += K^T x V,
        // reusing each V B-fragment for both WMMAs.
        const v16h ktA0 = load_A(Kts, CC, wv * 16, 0);
        const v16h ktA1 = load_A(Kts, CC, wv * 16, 32);
        const v16h asA0 = load_A(As,  CC, wv * 16, 0);
        const v16h asA1 = load_A(As,  CC, wv * 16, 32);   // zero-region for wv<2 (still correct)
#pragma unroll
        for (int j = 0; j < 4; ++j) {
            v8f acc = {};
            acc = __builtin_amdgcn_wmma_f32_16x16x32_f16(
                      false, qA0, false, load_Bt(Sft, EDIM, 0,  j * 16), (short)0, acc, false, false);
            acc = __builtin_amdgcn_wmma_f32_16x16x32_f16(
                      false, qA1, false, load_Bt(Sft, EDIM, 32, j * 16), (short)0, acc, false, false);

            const v16h vB0 = load_Bt(Vts, CC, 0,  j * 16);
            const v16h vB1 = load_Bt(Vts, CC, 32, j * 16);

            acc = __builtin_amdgcn_wmma_f32_16x16x32_f16(
                      false, asA0, false, vB0, (short)0, acc, false, false);
            if (wv >= 2)   // attn cols 32..63 are all-zero for bands 0,1
                acc = __builtin_amdgcn_wmma_f32_16x16x32_f16(
                          false, asA1, false, vB1, (short)0, acc, false, false);

            Sreg[j] = __builtin_amdgcn_wmma_f32_16x16x32_f16(
                          false, ktA0, false, vB0, (short)0, Sreg[j], false, false);
            Sreg[j] = __builtin_amdgcn_wmma_f32_16x16x32_f16(
                          false, ktA1, false, vB1, (short)0, Sreg[j], false, false);

#pragma unroll
            for (int v = 0; v < 8; ++v) {
                const int row = wv * 16 + mb + v;
                const size_t o = ((size_t)(nb * LSEQ + l0 + row) * NH + hh) * EDIM
                               + j * 16 + nlane;
                Og[o] = acc[v] * zbuf[row];
            }
        }
        __syncthreads();   // protect LDS reuse across chunk iterations
    }
}

extern "C" void kernel_launch(void* const* d_in, const int* in_sizes, int n_in,
                              void* d_out, int out_size, void* d_ws, size_t ws_size,
                              hipStream_t stream) {
    const float* q  = (const float*)d_in[0];
    const float* k  = (const float*)d_in[1];
    const float* v  = (const float*)d_in[2];
    const float* kl = (const float*)d_in[3];
    float* out = (float*)d_out;
    (void)in_sizes; (void)n_in; (void)out_size; (void)d_ws; (void)ws_size;

    dim3 grid(NB * NH);   // one workgroup per (batch, head): sequential chunk scan inside
    dim3 block(128);      // 4 waves; each wave owns a 16-row band + a 16-row S band
    hipLaunchKernelGGL(causal_linear_attn_kernel, grid, block, 0, stream,
                       q, k, v, kl, out);
}
